// MemoryAugmentedAttention_77515569758928
// MI455X (gfx1250) — compile-verified, matrix operand order checked by reading, not run on hardware
//
#include <hip/hip_runtime.h>
#include <hip/hip_bf16.h>

// ---------------- problem constants ----------------
#define Hd   1024
#define NHh  16
#define HDd  64
#define MSs  256
#define MDd  512
#define Bb   2
#define Ss   2048
#define BS   (Bb*Ss)   // 4096 total rows

typedef __attribute__((ext_vector_type(16))) _Float16 v16h;
typedef __attribute__((ext_vector_type(8)))  _Float16 v8h;
typedef __attribute__((ext_vector_type(8)))  float    v8f;
typedef __attribute__((ext_vector_type(4)))  unsigned v4u;
typedef __attribute__((ext_vector_type(8)))  int      v8i;
typedef __attribute__((ext_vector_type(4)))  int      v4i;

union F16Frag { v16h v; v8h h[2]; };

#if defined(__has_builtin)
#if __has_builtin(__builtin_amdgcn_tensor_load_to_lds) && __has_builtin(__builtin_amdgcn_s_wait_tensorcnt)
#define HAVE_TDM 1
#endif
#endif
#ifndef HAVE_TDM
#define HAVE_TDM 0
#endif

// LDS byte offset of a __shared__ object: generic shared ptr = {SHARED_BASE, offset}
__device__ inline unsigned lds_off(const void* p) {
    return (unsigned)(unsigned long long)p;
}

// per-lane async 16B copy global -> LDS (ASYNCcnt)
__device__ inline void async_b128(unsigned lds_addr, const void* gp) {
    asm volatile("global_load_async_to_lds_b128 %0, %1, off"
                 :: "v"(lds_addr), "v"((unsigned long long)gp)
                 : "memory");
}

#if HAVE_TDM
// TDM: 2D tile DMA global->LDS. tile = rows x 32 f16 elems, row stride = strideElems.
// LDS dest: rows stored back-to-back, 64B per row (matches our tile layout exactly).
__device__ inline void tdm_load_tile(unsigned lds_addr, const void* gp,
                                     unsigned rows, unsigned strideElems) {
    unsigned long long ga = (unsigned long long)gp;
    v4u g0;
    g0[0] = 1u;                                    // count=1, is_restore=0, gather=0
    g0[1] = lds_addr;                              // lds_addr[31:0]
    g0[2] = (unsigned)(ga & 0xffffffffu);          // global_addr[31:0]
    g0[3] = (unsigned)(((ga >> 32) & 0x01ffffffu)  // global_addr[56:32]
                       | (2u << 30));              // type=2 (image)
    v8i g1;
    g1[0] = (int)(1u << 16);                       // data_size=1 (2 bytes), no pad/iter/mask
    g1[1] = (int)((32u & 0xffffu) << 16);          // tensor_dim0 = 32 (lo16 in [31:16])
    g1[2] = (int)(((32u >> 16) & 0xffffu) | ((rows & 0xffffu) << 16)); // dim0 hi, tensor_dim1 lo
    g1[3] = (int)(((rows >> 16) & 0xffffu) | (32u << 16));             // dim1 hi, tile_dim0=32
    g1[4] = (int)(rows & 0xffffu);                 // tile_dim1 = rows, tile_dim2 = 0
    g1[5] = (int)(strideElems);                    // tensor_dim0_stride lo32
    g1[6] = 0;                                     // stride hi16, dim1_stride lo16
    g1[7] = 0;
    v4i z4 = {0, 0, 0, 0};
    v8i z8 = {0, 0, 0, 0, 0, 0, 0, 0};
    __builtin_amdgcn_tensor_load_to_lds(g0, g1, z4, z4, z8, 0);
}
#endif

// ---------------- elementwise converts ----------------
__global__ __launch_bounds__(256)
void f32_to_f16_kernel(const float* __restrict__ in, _Float16* __restrict__ out, int n) {
    int i = blockIdx.x * 256 + threadIdx.x;
    if (i < n) out[i] = (_Float16)in[i];
}

// memory_values [MS,MD] f32 -> transposed f16 [MD,MS]
__global__ __launch_bounds__(256)
void transpose_mv_kernel(const float* __restrict__ in, _Float16* __restrict__ out) {
    int i = blockIdx.x * 256 + threadIdx.x;
    if (i < MSs * MDd) {
        int r = i / MDd, c = i % MDd;
        out[c * MSs + r] = (_Float16)in[i];
    }
}

// ---------------- row softmax: f32 [rows,256] -> f16 probs ----------------
__global__ __launch_bounds__(256)
void softmax256_kernel(const float* __restrict__ in, _Float16* __restrict__ out) {
    int wid = threadIdx.x >> 5, lane = threadIdx.x & 31;
    int row = blockIdx.x * 8 + wid;
    const float* src = in + (size_t)row * MSs;
    float vv[8];
    #pragma unroll
    for (int i = 0; i < 8; ++i) vv[i] = src[lane + i * 32];
    float mx = vv[0];
    #pragma unroll
    for (int i = 1; i < 8; ++i) mx = fmaxf(mx, vv[i]);
    #pragma unroll
    for (int m = 1; m < 32; m <<= 1) mx = fmaxf(mx, __shfl_xor(mx, m, 32));
    float sum = 0.f;
    #pragma unroll
    for (int i = 0; i < 8; ++i) { vv[i] = __expf(vv[i] - mx); sum += vv[i]; }
    #pragma unroll
    for (int m = 1; m < 32; m <<= 1) sum += __shfl_xor(sum, m, 32);
    float inv = 1.0f / sum;
    _Float16* dst = out + (size_t)row * MSs;
    #pragma unroll
    for (int i = 0; i < 8; ++i) dst[lane + i * 32] = (_Float16)(vv[i] * inv);
}

// ---------------- WMMA GEMM: C[M,N] = scale*(A[M,K] @ W[N,K]^T + bias) ----------------
// Tiles staged via TDM tensor_load_to_lds (wave0 issues, TENSORcnt) with
// per-lane global_load_async_to_lds_b128 (ASYNCcnt) as fallback; double-buffered LDS.
// MODE 0: f32 out row-major; MODE 1: f16 out row-major; MODE 2: f16 out permuted [B,NH,S,HD]
template<int MODE>
__global__ __launch_bounds__(256)
void gemm_wmma_kernel(const _Float16* __restrict__ A, const _Float16* __restrict__ W,
                      const float* __restrict__ bias, void* __restrict__ outp,
                      int M, int N, int K, float scale)
{
    __shared__ __align__(16) _Float16 lA[2][128 * 32];
    __shared__ __align__(16) _Float16 lB[2][128 * 32];
    const int tid  = threadIdx.x;
    const int wid  = tid >> 5, lane = tid & 31;
    const int hsel = lane >> 4, r16 = lane & 15;
    const int wm = (wid >> 1) * 32;   // wave M offset within 128
    const int wn = (wid & 1) * 64;    // wave N offset within 128
    const int gm0 = blockIdx.y * 128;
    const int gn0 = blockIdx.x * 128;
    const int nT  = K >> 5;

    v8f acc[2][4];
    #pragma unroll
    for (int i = 0; i < 2; ++i)
        #pragma unroll
        for (int j = 0; j < 4; ++j) { v8f z = {}; acc[i][j] = z; }

    auto issue_tile = [&](int t) {
        const int k0 = t << 5;
        const int buf = t & 1;
#if HAVE_TDM
        if (wid == 0) {
            tdm_load_tile(lds_off(&lA[buf][0]), &A[(size_t)gm0 * K + k0], 128, (unsigned)K);
            tdm_load_tile(lds_off(&lB[buf][0]), &W[(size_t)gn0 * K + k0], 128, (unsigned)K);
        }
#else
        #pragma unroll
        for (int i = 0; i < 2; ++i) {
            int c = tid + i * 256;
            int row = c >> 2, cc = c & 3;
            async_b128(lds_off(&lA[buf][row * 32 + cc * 8]),
                       &A[(size_t)(gm0 + row) * K + k0 + cc * 8]);
            async_b128(lds_off(&lB[buf][row * 32 + cc * 8]),
                       &W[(size_t)(gn0 + row) * K + k0 + cc * 8]);
        }
#endif
    };
    auto wait_tile = [&](bool more_in_flight) {
#if HAVE_TDM
        if (wid == 0) {
            if (more_in_flight) __builtin_amdgcn_s_wait_tensorcnt((short)2);
            else                __builtin_amdgcn_s_wait_tensorcnt((short)0);
        }
#else
        if (more_in_flight) asm volatile("s_wait_asynccnt 0x4" ::: "memory");
        else                asm volatile("s_wait_asynccnt 0x0" ::: "memory");
#endif
    };

    issue_tile(0);
    for (int t = 0; t < nT; ++t) {
        if (t + 1 < nT) issue_tile(t + 1);       // prefetch into other buffer (DMA)
        wait_tile(t + 1 < nT);                   // tile t landed (counters are in-order)
        __syncthreads();

        const _Float16* bufA = lA[t & 1];
        const _Float16* bufB = lB[t & 1];
        F16Frag fa[2], fb[4];
        #pragma unroll
        for (int im = 0; im < 2; ++im) {
            int base = (wm + im * 16 + r16) * 32 + hsel * 8;
            fa[im].h[0] = *(const v8h*)&bufA[base];
            fa[im].h[1] = *(const v8h*)&bufA[base + 16];
        }
        #pragma unroll
        for (int jn = 0; jn < 4; ++jn) {
            int base = (wn + jn * 16 + r16) * 32 + hsel * 8;
            fb[jn].h[0] = *(const v8h*)&bufB[base];
            fb[jn].h[1] = *(const v8h*)&bufB[base + 16];
        }
        #pragma unroll
        for (int im = 0; im < 2; ++im)
            #pragma unroll
            for (int jn = 0; jn < 4; ++jn)
                acc[im][jn] = __builtin_amdgcn_wmma_f32_16x16x32_f16(
                    false, fa[im].v, false, fb[jn].v, (short)0, acc[im][jn], false, false);

        __syncthreads();   // done reading buf[t&1]; safe target for DMA at t+2
    }

    // epilogue (C-frag layout: vgpr r <-> row r + 8*hsel; lane%16 <-> col)
    #pragma unroll
    for (int im = 0; im < 2; ++im)
        #pragma unroll
        for (int jn = 0; jn < 4; ++jn) {
            int gc = gn0 + wn + jn * 16 + r16;
            float bv = bias ? bias[gc] : 0.0f;
            #pragma unroll
            for (int r = 0; r < 8; ++r) {
                int gr = gm0 + wm + im * 16 + r + 8 * hsel;
                float val = (acc[im][jn][r] + bv) * scale;
                if (MODE == 0) {
                    ((float*)outp)[(size_t)gr * N + gc] = val;
                } else if (MODE == 1) {
                    ((_Float16*)outp)[(size_t)gr * N + gc] = (_Float16)val;
                } else {
                    int b = gr >> 11, s = gr & (Ss - 1);
                    int hh = gc >> 6, hd = gc & (HDd - 1);
                    ((_Float16*)outp)[(((size_t)(b * NHh + hh)) * Ss + s) * HDd + hd] = (_Float16)val;
                }
            }
        }
}

// ---------------- flash attention over [B,NH,S,HD] f16 q/k/v ----------------
// block: 128 thr (4 waves), each wave owns 16 query rows; block owns 64.
// K-tile staged with global_load_async_to_lds_b128 (ASYNCcnt).
// epilogue fuses: comb = attn_out + memproj (memproj pre-scaled 0.3 + bias), f16 out.
__global__ __launch_bounds__(128)
void flash_attn_kernel(const _Float16* __restrict__ q, const _Float16* __restrict__ k,
                       const _Float16* __restrict__ v, const float* __restrict__ mask,
                       const float* __restrict__ memproj, _Float16* __restrict__ comb)
{
    __shared__ __align__(16) _Float16 lK[64 * HDd];        // [key][hd]
    __shared__ __align__(16) _Float16 lVT[HDd * 64];       // [hd][key]  (V^T)
    __shared__ __align__(16) _Float16 lP[4][16 * 64];      // per-wave P tile

    const int tid  = threadIdx.x;
    const int wid  = tid >> 5, lane = tid & 31;
    const int hsel = lane >> 4, r16 = lane & 15;
    const int b  = blockIdx.z, hh = blockIdx.y;
    const int q0 = blockIdx.x * 64;
    const size_t base = ((size_t)(b * NHh + hh)) * Ss * HDd;

    // Q fragments for this wave's 16 rows (K-dim = HD = 64 -> 2 k-steps)
    F16Frag fq[2];
    {
        const _Float16* qp = q + base + (size_t)(q0 + wid * 16 + r16) * HDd;
        #pragma unroll
        for (int kk = 0; kk < 2; ++kk) {
            int bk = kk * 32 + hsel * 8;
            fq[kk].h[0] = *(const v8h*)(qp + bk);
            fq[kk].h[1] = *(const v8h*)(qp + bk + 16);
        }
    }

    v8f o[4];
    #pragma unroll
    for (int j = 0; j < 4; ++j) { v8f z = {}; o[j] = z; }
    float m_s[8], l_s[8];
    #pragma unroll
    for (int r = 0; r < 8; ++r) { m_s[r] = -3.0e38f; l_s[r] = 0.0f; }

    const unsigned ldsK = lds_off(lK);

    for (int kt = 0; kt < Ss / 64; ++kt) {
        __syncthreads();                       // all waves done with previous K/V
        {   // K tile: contiguous 64*HD halfs = 8KB, async DMA to LDS
            const char* src = (const char*)(k + base + (size_t)kt * 64 * HDd);
            for (int i = tid; i < 512; i += 128)
                async_b128(ldsK + (unsigned)i * 16, src + (size_t)i * 16);
        }
        {   // V tile transposed into lVT[hd][key]
            const _Float16* vsrc = v + base + (size_t)kt * 64 * HDd;
            for (int i = tid; i < 4096; i += 128) {
                int key = i >> 6, hd = i & 63;
                lVT[hd * 64 + key] = vsrc[key * HDd + hd];
            }
        }
        asm volatile("s_wait_asynccnt 0x0" ::: "memory");
        __syncthreads();

        float mk[4];
        #pragma unroll
        for (int j = 0; j < 4; ++j)
            mk[j] = -1e9f * mask[(size_t)b * Ss + kt * 64 + j * 16 + r16];

        // scores: 16 x 64 via 4 n-tiles x 2 k-steps of WMMA
        v8f s[4];
        #pragma unroll
        for (int j = 0; j < 4; ++j) {
            v8f z = {}; s[j] = z;
            #pragma unroll
            for (int kk = 0; kk < 2; ++kk) {
                F16Frag fb;
                int bb = (j * 16 + r16) * HDd + kk * 32 + hsel * 8;
                fb.h[0] = *(const v8h*)&lK[bb];
                fb.h[1] = *(const v8h*)&lK[bb + 16];
                s[j] = __builtin_amdgcn_wmma_f32_16x16x32_f16(
                    false, fq[kk].v, false, fb.v, (short)0, s[j], false, false);
            }
            #pragma unroll
            for (int r = 0; r < 8; ++r) s[j][r] = s[j][r] * 0.125f + mk[j];
        }

        // online softmax; row r+8*hsel lives in vgpr r, spread over 16 lanes of the half
        #pragma unroll
        for (int r = 0; r < 8; ++r) {
            float mx = fmaxf(fmaxf(s[0][r], s[1][r]), fmaxf(s[2][r], s[3][r]));
            #pragma unroll
            for (int m = 1; m < 16; m <<= 1) mx = fmaxf(mx, __shfl_xor(mx, m, 32));
            float mnew = fmaxf(m_s[r], mx);
            float corr = __expf(m_s[r] - mnew);
            float psum = 0.f;
            #pragma unroll
            for (int j = 0; j < 4; ++j) {
                float p = __expf(s[j][r] - mnew);
                s[j][r] = p; psum += p;
            }
            #pragma unroll
            for (int m = 1; m < 16; m <<= 1) psum += __shfl_xor(psum, m, 32);
            l_s[r] = l_s[r] * corr + psum;
            m_s[r] = mnew;
            #pragma unroll
            for (int j = 0; j < 4; ++j) o[j][r] *= corr;
        }

        // P (C-layout f32) -> per-wave LDS as f16 A-layout source
        #pragma unroll
        for (int j = 0; j < 4; ++j)
            #pragma unroll
            for (int r = 0; r < 8; ++r)
                lP[wid][(r + 8 * hsel) * 64 + j * 16 + r16] = (_Float16)s[j][r];
        asm volatile("s_wait_dscnt 0x0" ::: "memory");  // same-wave DS RAW

        // O += P(16x64) @ V(64x64): B^T = V^T comes from lVT rows
        #pragma unroll
        for (int j2 = 0; j2 < 4; ++j2) {
            #pragma unroll
            for (int kk = 0; kk < 2; ++kk) {
                F16Frag fa, fb;
                int ab = r16 * 64 + kk * 32 + hsel * 8;
                fa.h[0] = *(const v8h*)&lP[wid][ab];
                fa.h[1] = *(const v8h*)&lP[wid][ab + 16];
                int bb = (j2 * 16 + r16) * 64 + kk * 32 + hsel * 8;
                fb.h[0] = *(const v8h*)&lVT[bb];
                fb.h[1] = *(const v8h*)&lVT[bb + 16];
                o[j2] = __builtin_amdgcn_wmma_f32_16x16x32_f16(
                    false, fa.v, false, fb.v, (short)0, o[j2], false, false);
            }
        }
    }

    // epilogue: normalize, add memory path, emit f16 combined [B,S,H]
    #pragma unroll
    for (int j2 = 0; j2 < 4; ++j2)
        #pragma unroll
        for (int r = 0; r < 8; ++r) {
            int gs = q0 + wid * 16 + r + 8 * hsel;
            size_t gi = ((size_t)b * Ss + gs) * Hd + hh * HDd + j2 * 16 + r16;
            comb[gi] = (_Float16)(o[j2][r] / l_s[r] + memproj[gi]);
        }
}

// ---------------- host launcher ----------------
extern "C" void kernel_launch(void* const* d_in, const int* in_sizes, int n_in,
                              void* d_out, int out_size, void* d_ws, size_t ws_size,
                              hipStream_t stream)
{
    const float* hs  = (const float*)d_in[0];
    const float* msk = (const float*)d_in[1];
    const float* Wq  = (const float*)d_in[2];
    const float* bq  = (const float*)d_in[3];
    const float* Wk  = (const float*)d_in[4];
    const float* bk  = (const float*)d_in[5];
    const float* Wv  = (const float*)d_in[6];
    const float* bv  = (const float*)d_in[7];
    const float* Wo  = (const float*)d_in[8];
    const float* bo  = (const float*)d_in[9];
    const float* mv  = (const float*)d_in[10];
    const float* Wmp = (const float*)d_in[11];
    const float* bmp = (const float*)d_in[12];
    const float* Wma = (const float*)d_in[13];
    const float* bma = (const float*)d_in[14];

    char* ws = (char*)d_ws;
    size_t off = 0;
    auto alloc = [&](size_t bytes) -> char* {
        char* p = ws + off; off += (bytes + 255) & ~(size_t)255; return p;
    };
    _Float16* hs_h   = (_Float16*)alloc((size_t)BS * Hd * 2);
    _Float16* wq_h   = (_Float16*)alloc((size_t)Hd * Hd * 2);
    _Float16* wk_h   = (_Float16*)alloc((size_t)Hd * Hd * 2);
    _Float16* wv_h   = (_Float16*)alloc((size_t)Hd * Hd * 2);
    _Float16* wo_h   = (_Float16*)alloc((size_t)Hd * Hd * 2);
    _Float16* wma_h  = (_Float16*)alloc((size_t)MSs * Hd * 2);
    _Float16* wmp_h  = (_Float16*)alloc((size_t)Hd * MDd * 2);
    _Float16* mvT_h  = (_Float16*)alloc((size_t)MDd * MSs * 2);
    _Float16* q_h    = (_Float16*)alloc((size_t)BS * Hd * 2);
    _Float16* k_h    = (_Float16*)alloc((size_t)BS * Hd * 2);
    _Float16* v_h    = (_Float16*)alloc((size_t)BS * Hd * 2);
    float*    msc_f  = (float*)   alloc((size_t)BS * MSs * 4);
    _Float16* mpr_h  = (_Float16*)alloc((size_t)BS * MSs * 2);
    _Float16* mout_h = (_Float16*)alloc((size_t)BS * MDd * 2);
    float*    mprj_f = (float*)   alloc((size_t)BS * Hd * 4);
    _Float16* comb_h = (_Float16*)alloc((size_t)BS * Hd * 2);

    auto cvt = [&](const float* in, _Float16* out, int n) {
        f32_to_f16_kernel<<<(n + 255) / 256, 256, 0, stream>>>(in, out, n);
    };
    cvt(hs,  hs_h,  BS * Hd);
    cvt(Wq,  wq_h,  Hd * Hd);
    cvt(Wk,  wk_h,  Hd * Hd);
    cvt(Wv,  wv_h,  Hd * Hd);
    cvt(Wo,  wo_h,  Hd * Hd);
    cvt(Wma, wma_h, MSs * Hd);
    cvt(Wmp, wmp_h, Hd * MDd);
    transpose_mv_kernel<<<(MSs * MDd + 255) / 256, 256, 0, stream>>>(mv, mvT_h);

    // Q/K/V projections -> f16 [B,NH,S,HD]
    dim3 gqkv(Hd / 128, BS / 128);
    gemm_wmma_kernel<2><<<gqkv, 256, 0, stream>>>(hs_h, wq_h, bq, q_h, BS, Hd, Hd, 1.0f);
    gemm_wmma_kernel<2><<<gqkv, 256, 0, stream>>>(hs_h, wk_h, bk, k_h, BS, Hd, Hd, 1.0f);
    gemm_wmma_kernel<2><<<gqkv, 256, 0, stream>>>(hs_h, wv_h, bv, v_h, BS, Hd, Hd, 1.0f);

    // memory path
    dim3 gms(MSs / 128, BS / 128);
    gemm_wmma_kernel<0><<<gms, 256, 0, stream>>>(hs_h, wma_h, bma, msc_f, BS, MSs, Hd, 1.0f);
    softmax256_kernel<<<BS / 8, 256, 0, stream>>>(msc_f, mpr_h);
    dim3 gmo(MDd / 128, BS / 128);
    gemm_wmma_kernel<1><<<gmo, 256, 0, stream>>>(mpr_h, mvT_h, nullptr, mout_h, BS, MDd, MSs, 1.0f);
    dim3 gmp(Hd / 128, BS / 128);
    gemm_wmma_kernel<0><<<gmp, 256, 0, stream>>>(mout_h, wmp_h, bmp, mprj_f, BS, Hd, MDd, 0.3f);

    // flash attention + fuse memory-path add -> f16 combined
    dim3 gfa(Ss / 64, NHh, Bb);
    flash_attn_kernel<<<gfa, 128, 0, stream>>>(q_h, k_h, v_h, msk, mprj_f, comb_h);

    // output projection -> f32 d_out
    dim3 gfin(Hd / 128, BS / 128);
    gemm_wmma_kernel<0><<<gfin, 256, 0, stream>>>(comb_h, wo_h, bo, (float*)d_out, BS, Hd, Hd, 1.0f);
}
